// MambaClassifier_24833500905596
// MI455X (gfx1250) — matrix operand order repaired and checked by reading, not compiled, not run on hardware
//
#include <hip/hip_runtime.h>
#include <stdint.h>

// ---------- model dims ----------
#define BB       8
#define LL       1024
#define TT       (BB*LL)        // 8192 tokens
#define IN_DIM   64
#define DMODEL   256
#define DINNER   512
#define DSTATE   16
#define DCONV    4
#define DTRANK   16
#define NLAYERS  2

// ---------- CDNA5 WMMA types ----------
typedef __attribute__((ext_vector_type(16))) __bf16 v16bf;
typedef __attribute__((ext_vector_type(8)))  float  v8f;
typedef uint32_t v4u __attribute__((ext_vector_type(4)));
typedef uint32_t v8u __attribute__((ext_vector_type(8)));

union FragBF {
  v16bf v;
  uint4 u[2];
};

__device__ __forceinline__ uint16_t f2bf(float f) {
  union { float f; uint32_t u; } c; c.f = f;
  uint32_t u = c.u;
  uint32_t r = (u + 0x7FFFu + ((u >> 16) & 1u)) >> 16;  // round-to-nearest-even
  return (uint16_t)r;
}

// ---------- CDNA5 async global->LDS helpers ----------
__device__ __forceinline__ uint32_t lds_off(const void* p) {
  // generic -> LDS(as3) addrspacecast; as3 pointers are 32-bit LDS byte offsets
  return (uint32_t)(uintptr_t)(__attribute__((address_space(3))) const void*)p;
}

__device__ __forceinline__ void async_b128(uint32_t lds_byte, const void* gptr) {
  // GLOBAL_LOAD_ASYNC_TO_LDS_B128: per-lane 16B global -> LDS, tracked by ASYNCcnt
  asm volatile("global_load_async_to_lds_b128 %0, %1, off"
               :: "v"(lds_byte), "v"(gptr) : "memory");
}

__device__ __forceinline__ void wait_async(int pending) {  // pending is wave-uniform
  if (pending) asm volatile("s_wait_asynccnt 4" ::: "memory");
  else         asm volatile("s_wait_asynccnt 0" ::: "memory");
}

__device__ __forceinline__ void wait_tensor(int pending) {
#if __has_builtin(__builtin_amdgcn_s_wait_tensorcnt)
  if (pending) __builtin_amdgcn_s_wait_tensorcnt(1);
  else         __builtin_amdgcn_s_wait_tensorcnt(0);
#else
  if (pending) asm volatile("s_wait_tensorcnt 1" ::: "memory");
  else         asm volatile("s_wait_tensorcnt 0" ::: "memory");
#endif
}

// TDM 2D tile load, D# groups per CDNA5 ISA (2-group form,
// VADDR2/3 = NULL -> tensors up to 2D). Inline asm bypasses builtin-arity skew.
__device__ __forceinline__ void tdm_load_2d(v4u g0, v8u g1) {
  asm volatile("tensor_load_to_lds %0, %1" :: "s"(g0), "s"(g1) : "memory");
}

// ---------- elementwise f32 -> bf16 ----------
__global__ void k_cvt_bf16(const float* __restrict__ src, uint16_t* __restrict__ dst, int n) {
  int i = blockIdx.x * blockDim.x + threadIdx.x;
  if (i < n) dst[i] = f2bf(src[i]);
}

// ============================================================================
// v1 GEMM (direct from global): C(MxN) = A(MxK,bf16) * W(NxK,bf16)^T (+bias)
// one 16x16 tile per wave. Used for the irregular shapes (K=64 / N=48).
// ============================================================================
__global__ void k_wmma_gemm(const uint16_t* __restrict__ A, int lda,
                            const uint16_t* __restrict__ W, int ldw,
                            float* __restrict__ C, int ldc,
                            const float* __restrict__ bias,
                            int M, int N, int K, int accumulate) {
  const int wave = threadIdx.x >> 5;
  const int lane = threadIdx.x & 31;
  const int tiles_n = N >> 4;
  const int tiles_total = (M >> 4) * tiles_n;
  const int tile = blockIdx.x * (blockDim.x >> 5) + wave;
  if (tile >= tiles_total) return;               // wave-uniform: EXEC all-1s at WMMA
  const int mt = tile / tiles_n;
  const int nt = tile - mt * tiles_n;
  const int row  = lane & 15;
  const int half = lane >> 4;

  const uint16_t* Ap = A + (size_t)(mt * 16 + row) * lda;
  const uint16_t* Wp = W + (size_t)(nt * 16 + row) * ldw;

  v8f acc = {};
  for (int kk = 0; kk < K; kk += 32) {
    FragBF a, b;
    a.u[0] = *reinterpret_cast<const uint4*>(Ap + kk + half * 8);
    a.u[1] = *reinterpret_cast<const uint4*>(Ap + kk + half * 8 + 16);
    b.u[0] = *reinterpret_cast<const uint4*>(Wp + kk + half * 8);
    b.u[1] = *reinterpret_cast<const uint4*>(Wp + kk + half * 8 + 16);
    if (kk + 32 < K) __builtin_prefetch(Wp + kk + 32, 0, 1);   // global_prefetch_b8
    acc = __builtin_amdgcn_wmma_f32_16x16x32_bf16(false, a.v, false, b.v,
                                                  (short)0, acc, false, false);
  }

  const float bv = bias ? bias[nt * 16 + row] : 0.0f;
  float* Cp = C + (size_t)(mt * 16 + half * 8) * ldc + nt * 16 + row;
#pragma unroll
  for (int i = 0; i < 8; i++) {
    float v = acc[i] + bv;
    if (accumulate) Cp[(size_t)i * ldc] += v;
    else            Cp[(size_t)i * ldc]  = v;
  }
}

// ============================================================================
// v2 GEMM (double-buffered async/TDM-LDS staged): 128-thread block -> 64x64
// output block. A panel staged with GLOBAL_LOAD_ASYNC_TO_LDS_B128 (ASYNCcnt),
// W panel staged with TENSOR_LOAD_TO_LDS (one 16x64 TDM tile per wave,
// TENSORcnt). Next K-chunk staged while current computes. Fragments read as
// ds_load_b128; each wave reuses its A fragment across 4 N-tiles.
// Requires M%64==0, N%64==0, K%64==0.
// ============================================================================
#define BM 64
#define BN 64
#define BKC 64

__global__ __launch_bounds__(128) void k_wmma_gemm_lds(
    const uint16_t* __restrict__ A, int lda,
    const uint16_t* __restrict__ W, int ldw,
    float* __restrict__ C, int ldc,
    int M, int N, int K, int accumulate) {
  __shared__ __align__(16) uint16_t As[2][BM * BKC];   // 2 x 8 KB
  __shared__ __align__(16) uint16_t Ws[2][BN * BKC];   // 2 x 8 KB
  const int tid  = threadIdx.x;
  const int lane = tid & 31;
  const int wv   = __builtin_amdgcn_readfirstlane(tid >> 5);  // wave id, SGPR
  const int nb   = N / BN;
  const int m0   = (blockIdx.x / nb) * BM;
  const int n0   = (blockIdx.x % nb) * BN;
  const int row  = lane & 15;
  const int half = lane >> 4;

  v8f acc[4] = {};   // this wave's 16(M) x 64(N) strip

  auto stage = [&](int buf, int kk) {
    // ---- A panel: 512 x 16B via per-lane async loads (4 per thread) ----
#pragma unroll
    for (int r = 0; r < 4; r++) {
      const int idx  = r * 128 + tid;        // [0,512) uint4 slots
      const int prow = idx >> 3;             // 8 x 16B per 64-elem row
      const int pcol = (idx & 7) * 8;        // bf16 column
      async_b128(lds_off(&As[buf][idx * 8]),
                 A + (size_t)(m0 + prow) * lda + kk + pcol);
    }
    // ---- W panel: each wave TDM-loads its 16-row quarter ----
    const uint64_t ga = (uint64_t)(uintptr_t)(W + (size_t)(n0 + wv * 16) * ldw + kk);
    const uint32_t la = lds_off(&Ws[buf][wv * 16 * BKC]);
    v4u g0; v8u g1;
    g0[0] = 1u;                                            // count=1, user D#
    g0[1] = la;                                            // lds_addr
    g0[2] = (uint32_t)ga;                                  // global_addr[31:0]
    g0[3] = ((uint32_t)(ga >> 32) & 0x01FFFFFFu) | 0x80000000u;  // addr[56:32]|type=2
    g1[0] = 1u << 16;                                      // data_size=1 (2 bytes)
    g1[1] = (BKC & 0xFFFFu) << 16;                         // tensor_dim0=64 lo16
    g1[2] = 16u << 16;                                     // dim0 hi=0 | tensor_dim1=16
    g1[3] = (uint32_t)BKC << 16;                           // dim1 hi=0 | tile_dim0=64
    g1[4] = 16u;                                           // tile_dim1=16, tile_dim2=0
    g1[5] = (uint32_t)ldw;                                 // tensor_dim0_stride lo32
    g1[6] = 0u;                                            // stride hi | dim1_stride lo
    g1[7] = 0u;
    tdm_load_2d(g0, g1);
  };

  const int nchunk = K / BKC;
  stage(0, 0);
  for (int c = 0; c < nchunk; c++) {
    const int  cur  = c & 1;
    const int  more = (c + 1 < nchunk);
    if (more) stage(cur ^ 1, (c + 1) * BKC);   // overlap next DMA with compute
    wait_async(more);                          // <=4 (next in flight) or 0
    wait_tensor(more);                         // <=1 (next in flight) or 0
    __syncthreads();                           // all quarters visible block-wide

#pragma unroll
    for (int ks = 0; ks < 2; ks++) {
      FragBF a;
      const uint16_t* ap = &As[cur][(wv * 16 + row) * BKC + ks * 32 + half * 8];
      a.u[0] = *reinterpret_cast<const uint4*>(ap);
      a.u[1] = *reinterpret_cast<const uint4*>(ap + 16);
#pragma unroll
      for (int nt = 0; nt < 4; nt++) {
        FragBF b;
        const uint16_t* wp = &Ws[cur][(nt * 16 + row) * BKC + ks * 32 + half * 8];
        b.u[0] = *reinterpret_cast<const uint4*>(wp);
        b.u[1] = *reinterpret_cast<const uint4*>(wp + 16);
        acc[nt] = __builtin_amdgcn_wmma_f32_16x16x32_bf16(false, a.v, false, b.v,
                                                          (short)0, acc[nt], false, false);
      }
    }
    __syncthreads();   // reads done before this buffer is restaged
  }

#pragma unroll
  for (int nt = 0; nt < 4; nt++) {
    float* Cp = C + (size_t)(m0 + wv * 16 + half * 8) * ldc + n0 + nt * 16 + row;
#pragma unroll
    for (int i = 0; i < 8; i++) {
      if (accumulate) Cp[(size_t)i * ldc] += acc[nt][i];
      else            Cp[(size_t)i * ldc]  = acc[nt][i];
    }
  }
}

// ---------- RMSNorm over DMODEL, write bf16 ----------
__global__ void k_rmsnorm_bf16(const float* __restrict__ h, const float* __restrict__ w,
                               uint16_t* __restrict__ out) {
  const int wave = threadIdx.x >> 5, lane = threadIdx.x & 31;
  const int tok  = blockIdx.x * (blockDim.x >> 5) + wave;
  if (tok >= TT) return;
  const float* hp = h + (size_t)tok * DMODEL;
  float v[8]; float s = 0.f;
#pragma unroll
  for (int j = 0; j < 8; j++) { v[j] = hp[lane + j * 32]; s += v[j] * v[j]; }
#pragma unroll
  for (int m = 16; m >= 1; m >>= 1) s += __shfl_xor(s, m, 32);
  const float rs = rsqrtf(s * (1.0f / DMODEL) + 1e-5f);
  uint16_t* op = out + (size_t)tok * DMODEL;
#pragma unroll
  for (int j = 0; j < 8; j++) op[lane + j * 32] = f2bf(v[j] * rs * w[lane + j * 32]);
}

// ---------- causal depthwise conv (width 4) + SiLU ----------
__global__ void k_conv_silu(const float* __restrict__ xz, const float* __restrict__ cw,
                            const float* __restrict__ cb, float* __restrict__ uf,
                            uint16_t* __restrict__ ubf) {
  const int g = blockIdx.x * blockDim.x + threadIdx.x;
  if (g >= TT * DINNER) return;
  const int t = g >> 9, d = g & 511;
  const int b = t >> 10, l = t & 1023;
  float acc = cb[d];
#pragma unroll
  for (int j = 0; j < DCONV; j++) {
    const int lt = l - (DCONV - 1) + j;
    if (lt >= 0) acc += cw[d * DCONV + j] * xz[((size_t)(b * LL + lt)) * (2 * DINNER) + d];
  }
  const float s = acc / (1.0f + __expf(-acc));
  uf[g] = s;
  ubf[g] = f2bf(s);
}

// ---------- dt_proj + softplus ----------
__global__ void k_dt_softplus(const float* __restrict__ dbc, const float* __restrict__ dtw,
                              const float* __restrict__ dtb, float* __restrict__ delta) {
  const int g = blockIdx.x * blockDim.x + threadIdx.x;
  if (g >= TT * DINNER) return;
  const int t = g >> 9, d = g & 511;
  float acc = dtb[d];
  const float* r  = dbc + (size_t)t * 48;
  const float* wv = dtw + (size_t)d * DTRANK;
#pragma unroll
  for (int j = 0; j < DTRANK; j++) acc += r[j] * wv[j];
  delta[g] = (acc > 20.0f) ? acc : log1pf(__expf(acc));
}

// ---------- selective scan: one lane per (b,d) channel, 16-state in VGPRs ----------
__global__ void k_scan(const float* __restrict__ delta, const float* __restrict__ dbc,
                       const float* __restrict__ A_log, const float* __restrict__ Dpm,
                       float* __restrict__ uy) {
  const int g = blockIdx.x * blockDim.x + threadIdx.x;
  if (g >= BB * DINNER) return;
  const int b = g >> 9, d = g & 511;
  float A[DSTATE], h[DSTATE];
#pragma unroll
  for (int s = 0; s < DSTATE; s++) { A[s] = -__expf(A_log[d * DSTATE + s]); h[s] = 0.f; }
  const float Dv = Dpm[d];
  for (int l = 0; l < LL; l++) {
    const size_t t = (size_t)b * LL + l;
    const float dl = delta[t * DINNER + d];
    const float u  = uy[t * DINNER + d];
    const float* bc = dbc + t * 48;          // [0:16)=dt, [16:32)=B, [32:48)=C
    const float du = dl * u;
    float y = 0.f;
#pragma unroll
    for (int s = 0; s < DSTATE; s++) {
      h[s] = __expf(dl * A[s]) * h[s] + du * bc[16 + s];
      y += h[s] * bc[32 + s];
    }
    uy[t * DINNER + d] = y + Dv * u;
  }
}

// ---------- y * silu(z), write bf16 ----------
__global__ void k_gate(const float* __restrict__ y, const float* __restrict__ xz,
                       uint16_t* __restrict__ ybf) {
  const int g = blockIdx.x * blockDim.x + threadIdx.x;
  if (g >= TT * DINNER) return;
  const int t = g >> 9, d = g & 511;
  const float z = xz[(size_t)t * (2 * DINNER) + DINNER + d];
  const float sz = z / (1.0f + __expf(-z));
  ybf[g] = f2bf(y[g] * sz);
}

// ---------- classifier head on last token of each batch ----------
__global__ void k_head(const float* __restrict__ h, const float* __restrict__ w1,
                       const float* __restrict__ b1, const float* __restrict__ w2,
                       const float* __restrict__ b2, float* __restrict__ out) {
  __shared__ float h1[BB * 32];
  const int tid = threadIdx.x;                  // 256 threads
  const int tok = tid >> 5, unit = tid & 31;
  const float* hp = h + ((size_t)tok * LL + (LL - 1)) * DMODEL;
  float acc = b1[unit];
  for (int k = 0; k < DMODEL; k++) acc += hp[k] * w1[unit * DMODEL + k];
  h1[tid] = fmaxf(acc, 0.0f);
  __syncthreads();
  if (tid < BB) {
    float o = b2[0];
    for (int u2 = 0; u2 < 32; u2++) o += h1[tid * 32 + u2] * w2[u2];
    out[tid] = o;
  }
}

// ---------- workspace layout (bytes) ----------
static constexpr size_t OFF_H     = 0;
static constexpr size_t OFF_NBUF  = OFF_H     + (size_t)TT * DMODEL * 4;
static constexpr size_t OFF_XZ    = OFF_NBUF  + (size_t)TT * DMODEL * 2;
static constexpr size_t OFF_UF    = OFF_XZ    + (size_t)TT * 2 * DINNER * 4;
static constexpr size_t OFF_UBF   = OFF_UF    + (size_t)TT * DINNER * 4;
static constexpr size_t OFF_DBC   = OFF_UBF   + (size_t)TT * DINNER * 2;
static constexpr size_t OFF_DELTA = OFF_DBC   + (size_t)TT * 48 * 4;
static constexpr size_t OFF_YBF   = OFF_DELTA + (size_t)TT * DINNER * 4;
static constexpr size_t OFF_XBF   = OFF_YBF   + (size_t)TT * DINNER * 2;
static constexpr size_t OFF_EMBW  = OFF_XBF   + (size_t)TT * IN_DIM * 2;
static constexpr size_t OFF_IPW   = OFF_EMBW  + (size_t)DMODEL * IN_DIM * 2;
static constexpr size_t OFF_XPW   = OFF_IPW   + (size_t)NLAYERS * 2 * DINNER * DMODEL * 2;
static constexpr size_t OFF_OPW   = OFF_XPW   + (size_t)NLAYERS * 48 * DINNER * 2;

static inline void gemm_v1(const uint16_t* A, int lda, const uint16_t* W, int ldw,
                           float* C, int ldc, const float* bias,
                           int M, int N, int K, int acc, hipStream_t s) {
  const int tiles = (M / 16) * (N / 16);
  const int blocks = (tiles + 3) / 4;           // 4 waves per 128-thread block
  k_wmma_gemm<<<blocks, 128, 0, s>>>(A, lda, W, ldw, C, ldc, bias, M, N, K, acc);
}

static inline void gemm_lds(const uint16_t* A, int lda, const uint16_t* W, int ldw,
                            float* C, int ldc, int M, int N, int K, int acc, hipStream_t s) {
  k_wmma_gemm_lds<<<(M / BM) * (N / BN), 128, 0, s>>>(A, lda, W, ldw, C, ldc, M, N, K, acc);
}

extern "C" void kernel_launch(void* const* d_in, const int* in_sizes, int n_in,
                              void* d_out, int out_size, void* d_ws, size_t ws_size,
                              hipStream_t stream) {
  const float* x       = (const float*)d_in[0];
  const float* emb_w   = (const float*)d_in[1];
  const float* emb_b   = (const float*)d_in[2];
  const float* norm_w  = (const float*)d_in[3];
  const float* in_proj = (const float*)d_in[4];
  const float* conv_w  = (const float*)d_in[5];
  const float* conv_b  = (const float*)d_in[6];
  const float* x_proj  = (const float*)d_in[7];
  const float* dt_w    = (const float*)d_in[8];
  const float* dt_b    = (const float*)d_in[9];
  const float* A_log   = (const float*)d_in[10];
  const float* Dp      = (const float*)d_in[11];
  const float* out_w   = (const float*)d_in[12];
  const float* hw1     = (const float*)d_in[13];
  const float* hb1     = (const float*)d_in[14];
  const float* hw2     = (const float*)d_in[15];
  const float* hb2     = (const float*)d_in[16];

  char* ws = (char*)d_ws;
  float*    hbuf  = (float*)   (ws + OFF_H);
  uint16_t* nbuf  = (uint16_t*)(ws + OFF_NBUF);
  float*    xzbuf = (float*)   (ws + OFF_XZ);
  float*    ufbuf = (float*)   (ws + OFF_UF);
  uint16_t* ubf   = (uint16_t*)(ws + OFF_UBF);
  float*    dbc   = (float*)   (ws + OFF_DBC);
  float*    delta = (float*)   (ws + OFF_DELTA);
  uint16_t* ybf   = (uint16_t*)(ws + OFF_YBF);
  uint16_t* xbf   = (uint16_t*)(ws + OFF_XBF);
  uint16_t* embw  = (uint16_t*)(ws + OFF_EMBW);
  uint16_t* ipw   = (uint16_t*)(ws + OFF_IPW);
  uint16_t* xpw   = (uint16_t*)(ws + OFF_XPW);
  uint16_t* opw   = (uint16_t*)(ws + OFF_OPW);

  auto cvt = [&](const float* s, uint16_t* d, int n) {
    k_cvt_bf16<<<(n + 255) / 256, 256, 0, stream>>>(s, d, n);
  };

  // weight / input bf16 conversion
  cvt(x,       xbf,  TT * IN_DIM);
  cvt(emb_w,   embw, DMODEL * IN_DIM);
  cvt(in_proj, ipw,  NLAYERS * 2 * DINNER * DMODEL);
  cvt(x_proj,  xpw,  NLAYERS * 48 * DINNER);
  cvt(out_w,   opw,  NLAYERS * DMODEL * DINNER);

  // embedding: h = x @ emb_w^T + emb_b   (K=64 -> direct kernel)
  gemm_v1(xbf, IN_DIM, embw, IN_DIM, hbuf, DMODEL, emb_b, TT, DMODEL, IN_DIM, 0, stream);

  const int nEW = TT * DINNER;
  for (int l = 0; l < NLAYERS; l++) {
    // rmsnorm -> bf16
    k_rmsnorm_bf16<<<TT / 8, 256, 0, stream>>>(hbuf, norm_w + l * DMODEL, nbuf);
    // in_proj: xz = n @ W^T  (N=1024, K=256) -> double-buffered async/TDM WMMA
    gemm_lds(nbuf, DMODEL, ipw + (size_t)l * 2 * DINNER * DMODEL, DMODEL,
             xzbuf, 2 * DINNER, TT, 2 * DINNER, DMODEL, 0, stream);
    // depthwise conv + silu
    k_conv_silu<<<(nEW + 255) / 256, 256, 0, stream>>>(
        xzbuf, conv_w + (size_t)l * DINNER * DCONV, conv_b + l * DINNER, ufbuf, ubf);
    // x_proj: dbc = u @ W^T (N=48, K=512) -> direct kernel (N not mult of 64)
    gemm_v1(ubf, DINNER, xpw + (size_t)l * 48 * DINNER, DINNER,
            dbc, 48, nullptr, TT, 48, DINNER, 0, stream);
    // dt_proj + softplus
    k_dt_softplus<<<(nEW + 255) / 256, 256, 0, stream>>>(
        dbc, dt_w + (size_t)l * DINNER * DTRANK, dt_b + l * DINNER, delta);
    // selective scan (u -> y in place)
    k_scan<<<(BB * DINNER) / 256, 256, 0, stream>>>(
        delta, dbc, A_log + (size_t)l * DINNER * DSTATE, Dp + l * DINNER, ufbuf);
    // gate: ybf = bf16(y * silu(z))
    k_gate<<<(nEW + 255) / 256, 256, 0, stream>>>(ufbuf, xzbuf, ybf);
    // out_proj with residual: h += y @ W^T  (N=256, K=512) -> double-buffered WMMA
    gemm_lds(ybf, DINNER, opw + (size_t)l * DMODEL * DINNER, DINNER,
             hbuf, DMODEL, TT, DMODEL, DINNER, 1, stream);
  }

  // head
  k_head<<<1, 256, 0, stream>>>(hbuf, hw1, hb1, hw2, hb2, (float*)d_out);
}
